// MSA_8246337208414
// MI455X (gfx1250) — matrix operand order repaired and checked by reading, not compiled, not run on hardware
//
#include <hip/hip_runtime.h>
#include <stdint.h>

// Problem constants (from reference): B=8, S=1024, HID=1024, H=16, D=64
#define B_   8
#define S_   1024
#define HID_ 1024
#define H_   16
#define D_   64

typedef __bf16 bf16;
typedef __attribute__((ext_vector_type(16))) __bf16 v16bf;
typedef __attribute__((ext_vector_type(8)))  __bf16 v8bf;
typedef __attribute__((ext_vector_type(4)))  __bf16 v4bf;
typedef __attribute__((ext_vector_type(8)))  float  v8f;
typedef __attribute__((ext_vector_type(4)))  float  v4f;

static_assert(sizeof(bf16) == 2, "bf16 must be 2 bytes");

union V16U { v16bf v; v8bf h[2]; };

__device__ __forceinline__ v8f wmma_bf16(v16bf a, v16bf b, v8f c) {
  // D = A(16x32 bf16) * B(32x16 bf16) + C(16x16 f32)
  return __builtin_amdgcn_wmma_f32_16x16x32_bf16(
      /*neg_a=*/false, a, /*neg_b=*/false, b,
      /*c_mod=*/(short)0, c, /*reuse_a=*/false, /*reuse_b=*/false);
}

// Async global -> LDS copy, 16 bytes per lane, tracked by ASYNCcnt.
// LDS operand: low 32 bits of a generic LDS pointer are the DS address.
__device__ __forceinline__ void async_copy_b128(const bf16* gsrc, bf16* ldst) {
  const uint32_t lds = (uint32_t)(uintptr_t)ldst;
  asm volatile("global_load_async_to_lds_b128 %0, %1, off"
               :: "v"(lds), "v"(gsrc) : "memory");
}
__device__ __forceinline__ void wait_async0() {
  asm volatile("s_wait_asynccnt 0" ::: "memory");
}

// log2(e) / sqrt(D) = 1.4426950408889634 / 8, folded into Q so softmax uses exp2
#define QSCALE 0.18033688011112042f

// ---------------------------------------------------------------------------
// Kernel 1: per-head QKV projection, fp32 in -> bf16 out (Q pre-scaled).
// Block: 256 threads (8 waves). Each block: 128 rows x 1 head.
// ---------------------------------------------------------------------------
__global__ __launch_bounds__(256) void qkv_kernel(
    const float* __restrict__ x,
    const float* __restrict__ Wq, const float* __restrict__ bq,
    const float* __restrict__ Wk, const float* __restrict__ bk,
    const float* __restrict__ Wv, const float* __restrict__ bv,
    bf16* __restrict__ Qo, bf16* __restrict__ Ko, bf16* __restrict__ Vo)
{
  __shared__ bf16  sX[128][72];       // X tile (bf16), padded rows
  __shared__ bf16  sW[3][64][72];     // Wq/Wk/Wv for this head (bf16)
  __shared__ float sB[3][64];         // biases

  const int tid  = threadIdx.x;
  const int h    = blockIdx.y;
  const int row0 = blockIdx.x * 128;  // flat row in [0, B*S)

  // Cooperative load: X tile 128x64 fp32 -> bf16 LDS
  for (int i = tid; i < 128 * 16; i += 256) {
    const int r = i >> 4, cg = (i & 15) << 2;
    v4f xv = *(const v4f*)(x + (size_t)(row0 + r) * HID_ + h * D_ + cg);
    v4bf bb;
    bb[0] = (bf16)xv[0]; bb[1] = (bf16)xv[1];
    bb[2] = (bf16)xv[2]; bb[3] = (bf16)xv[3];
    *(v4bf*)&sX[r][cg] = bb;
  }
  // Cooperative load: weights 3 x 64x64 fp32 -> bf16 LDS
  {
    const float* Ws[3] = {Wq + h * D_ * D_, Wk + h * D_ * D_, Wv + h * D_ * D_};
    #pragma unroll
    for (int mm = 0; mm < 3; ++mm) {
      for (int i = tid; i < 64 * 16; i += 256) {
        const int r = i >> 4, cg = (i & 15) << 2;
        v4f wv = *(const v4f*)(Ws[mm] + r * D_ + cg);
        v4bf bb;
        bb[0] = (bf16)wv[0]; bb[1] = (bf16)wv[1];
        bb[2] = (bf16)wv[2]; bb[3] = (bf16)wv[3];
        *(v4bf*)&sW[mm][r][cg] = bb;
      }
    }
    if (tid < 192) {
      const float* Bs[3] = {bq, bk, bv};
      sB[tid >> 6][tid & 63] = Bs[tid >> 6][h * D_ + (tid & 63)];
    }
  }
  __syncthreads();

  const int wave = tid >> 5, lane = tid & 31;
  const int m = lane & 15, g = lane >> 4;   // C/D: col = m, row-half = g
  const int rb = wave * 16;

  // A fragments (16x32 bf16, two K-chunks covering K=64).
  // A layout: lane m holds row m; elem e -> K = (e%8) + 8*((e/8)*2 + g) + 32c
  V16U a[2];
  #pragma unroll
  for (int c = 0; c < 2; ++c) {
    const int d0 = c * 32 + g * 8;
    a[c].h[0] = *(const v8bf*)&sX[rb + m][d0];
    a[c].h[1] = *(const v8bf*)&sX[rb + m][d0 + 16];
  }

  bf16* Outs[3] = {Qo, Ko, Vo};
  #pragma unroll
  for (int mm = 0; mm < 3; ++mm) {
    #pragma unroll
    for (int t = 0; t < 4; ++t) {     // 4 tiles of 16 output features
      v8f acc = {};
      #pragma unroll
      for (int c = 0; c < 2; ++c) {
        // B layout: lane -> col n = t*16+m (= W row); elem e -> K = (e%8) + 8*((e/8)*2 + g) + 32c
        V16U bm;
        const int d0 = c * 32 + g * 8;
        bm.h[0] = *(const v8bf*)&sW[mm][t * 16 + m][d0];
        bm.h[1] = *(const v8bf*)&sW[mm][t * 16 + m][d0 + 16];
        acc = wmma_bf16(a[c].v, bm.v, acc);
      }
      const float bias = sB[mm][t * 16 + m];
      #pragma unroll
      for (int r = 0; r < 8; ++r) {   // C/D: vgpr r -> row r + 8g
        const int grow = row0 + rb + r + 8 * g;
        const int bb = grow >> 10;    // batch (S_ == 1024)
        const int ss = grow & 1023;   // seq position
        float val = acc[r] + bias;
        if (mm == 0) val *= QSCALE;
        Outs[mm][(((size_t)bb * H_ + h) * S_ + ss) * D_ + t * 16 + m] = (bf16)val;
      }
    }
  }
}

// ---------------------------------------------------------------------------
// Kernel 2: flash attention. Block: 256 threads = 8 waves = 128 query rows
// of one (b,h). Loops over 8 key-blocks of 128 keys. K tile arrives via
// async global->LDS DMA (ASYNCcnt); V is transposed into LDS through VGPRs.
// ---------------------------------------------------------------------------
__global__ __launch_bounds__(256) void attn_kernel(
    const bf16* __restrict__ Q, const bf16* __restrict__ K,
    const bf16* __restrict__ V, float* __restrict__ out)
{
  __shared__ bf16 sK[128][72];        // K tile, row-major [key][d]
  __shared__ bf16 sVT[64][136];       // V tile transposed [d][key]
  __shared__ bf16 sP[8][16][136];     // per-wave P staging [row][key]

  const int tid = threadIdx.x;
  const int bh  = blockIdx.y;
  const int b   = bh >> 4, h = bh & 15;
  const int q0  = blockIdx.x * 128;
  const size_t hb = (size_t)bh * S_ * D_;

  const int wave = tid >> 5, lane = tid & 31;
  const int m = lane & 15, g = lane >> 4;

  // Q A-fragments (held in registers for the whole kernel)
  V16U qa[2];
  {
    const bf16* qrow = Q + hb + (size_t)(q0 + wave * 16 + m) * D_;
    #pragma unroll
    for (int c = 0; c < 2; ++c) {
      const int d0 = c * 32 + g * 8;
      qa[c].h[0] = *(const v8bf*)(qrow + d0);
      qa[c].h[1] = *(const v8bf*)(qrow + d0 + 16);
    }
  }

  v8f o[4] = {};                       // O accumulator: 16 rows x 64 cols
  float mrow[8], lrow[8];
  #pragma unroll
  for (int r = 0; r < 8; ++r) { mrow[r] = -3.0e38f; lrow[r] = 0.0f; }

  for (int kb = 0; kb < S_ / 128; ++kb) {
    const int k0 = kb * 128;
    __syncthreads();                   // previous tile fully consumed

    // K tile: async DMA global -> LDS, 16B per lane, no VGPR staging
    #pragma unroll
    for (int i = 0; i < 4; ++i) {
      const int idx = tid + i * 256;          // 1024 chunks of 8 bf16
      const int r = idx >> 3, cg = (idx & 7) << 3;
      async_copy_b128(K + hb + (size_t)(k0 + r) * D_ + cg, &sK[r][cg]);
    }
    // V tile: load through VGPRs and transpose into LDS
    #pragma unroll
    for (int i = 0; i < 4; ++i) {
      const int idx = tid + i * 256;
      const int r = idx >> 3, cg = (idx & 7) << 3;
      v8bf vv = *(const v8bf*)(V + hb + (size_t)(k0 + r) * D_ + cg);
      #pragma unroll
      for (int j = 0; j < 8; ++j) sVT[cg + j][r] = vv[j];
    }
    if (kb + 1 < S_ / 128) {           // warm L2 for next tile -> global_prefetch_b8
      __builtin_prefetch(K + hb + (size_t)(k0 + 128) * D_ + (tid << 2), 0, 1);
      __builtin_prefetch(V + hb + (size_t)(k0 + 128) * D_ + (tid << 2), 0, 1);
    }
    wait_async0();                     // K tile landed in LDS
    __syncthreads();

    // S = Q * K^T  (Q carries log2e/sqrt(D))
    v8f st[8];
    #pragma unroll
    for (int t = 0; t < 8; ++t) {
      v8f acc = {};
      #pragma unroll
      for (int c = 0; c < 2; ++c) {
        V16U bm;
        const int d0 = c * 32 + g * 8;
        bm.h[0] = *(const v8bf*)&sK[t * 16 + m][d0];
        bm.h[1] = *(const v8bf*)&sK[t * 16 + m][d0 + 16];
        acc = wmma_bf16(qa[c].v, bm.v, acc);
      }
      st[t] = acc;
    }

    // Online softmax. C-layout: vgpr r -> query row r+8g; lane m -> key col.
    #pragma unroll
    for (int r = 0; r < 8; ++r) {
      float v = st[0][r];
      #pragma unroll
      for (int t = 1; t < 8; ++t) v = fmaxf(v, st[t][r]);
      #pragma unroll
      for (int off = 1; off < 16; off <<= 1)
        v = fmaxf(v, __shfl_xor(v, off, 32));
      const float mn = fmaxf(mrow[r], v);
      const float sc = __builtin_amdgcn_exp2f(mrow[r] - mn);
      mrow[r] = mn;
      lrow[r] *= sc;
      o[0][r] *= sc; o[1][r] *= sc; o[2][r] *= sc; o[3][r] *= sc;
      float rs = 0.0f;
      #pragma unroll
      for (int t = 0; t < 8; ++t) {
        const float p = __builtin_amdgcn_exp2f(st[t][r] - mn);
        st[t][r] = p;
        rs += p;
      }
      #pragma unroll
      for (int off = 1; off < 16; off <<= 1)
        rs += __shfl_xor(rs, off, 32);
      lrow[r] += rs;
    }

    // P: C-layout f32 -> A-layout bf16 via per-wave LDS staging
    #pragma unroll
    for (int t = 0; t < 8; ++t)
      #pragma unroll
      for (int r = 0; r < 8; ++r)
        sP[wave][r + 8 * g][t * 16 + m] = (bf16)st[t][r];

    // O += P * V   (4 K-chunks of 32 keys, 4 tiles of 16 dims)
    #pragma unroll
    for (int c = 0; c < 4; ++c) {
      V16U pa;
      const int d0 = c * 32 + g * 8;
      pa.h[0] = *(const v8bf*)&sP[wave][m][d0];
      pa.h[1] = *(const v8bf*)&sP[wave][m][d0 + 16];
      #pragma unroll
      for (int t = 0; t < 4; ++t) {
        V16U bm;
        const int kk = c * 32 + g * 8;
        bm.h[0] = *(const v8bf*)&sVT[t * 16 + m][kk];
        bm.h[1] = *(const v8bf*)&sVT[t * 16 + m][kk + 16];
        o[t] = wmma_bf16(pa.v, bm.v, o[t]);
      }
    }
  }

  // Epilogue: O / l -> fp32 output [B,S,HID]
  #pragma unroll
  for (int r = 0; r < 8; ++r) {
    const float inv = 1.0f / lrow[r];
    const int srow = q0 + wave * 16 + r + 8 * g;
    float* op = out + ((size_t)b * S_ + srow) * HID_ + h * D_ + m;
    #pragma unroll
    for (int t = 0; t < 4; ++t)
      op[t * 16] = o[t][r] * inv;
  }
}

// ---------------------------------------------------------------------------
extern "C" void kernel_launch(void* const* d_in, const int* in_sizes, int n_in,
                              void* d_out, int out_size, void* d_ws, size_t ws_size,
                              hipStream_t stream) {
  (void)in_sizes; (void)n_in; (void)out_size; (void)ws_size;
  const float* x  = (const float*)d_in[0];
  const float* Wq = (const float*)d_in[1];
  const float* bq = (const float*)d_in[2];
  const float* Wk = (const float*)d_in[3];
  const float* bk = (const float*)d_in[4];
  const float* Wv = (const float*)d_in[5];
  const float* bv = (const float*)d_in[6];
  float* out = (float*)d_out;

  // Workspace: Q,K,V bf16 [B,H,S,D] = 16 MB each (48 MB total)
  bf16* Qw = (bf16*)d_ws;
  const size_t NQ = (size_t)B_ * H_ * S_ * D_;
  bf16* Kw = Qw + NQ;
  bf16* Vw = Kw + NQ;

  qkv_kernel<<<dim3((B_ * S_) / 128, H_), 256, 0, stream>>>(
      x, Wq, bq, Wk, bk, Wv, bv, Qw, Kw, Vw);
  attn_kernel<<<dim3(S_ / 128, B_ * H_), 256, 0, stream>>>(Qw, Kw, Vw, out);
}
